// AttnModel_53824530153550
// MI455X (gfx1250) — matrix operand order repaired and checked by reading, not compiled, not run on hardware
//
#include <hip/hip_runtime.h>
#include <hip/hip_bf16.h>

// ---------------- problem constants ----------------
#define FEAT   128
#define EDGEF  128
#define TIMEF  128
#define DMDIM  384      // FEAT + EDGE + TIME
#define NHEAD  2
#define DKDIM  192      // DM / H
#define BATCH  512
#define NSRC   16
#define NNTOT  512
#define NNGH   32       // neighbors per source
#define NBLK   128      // threads per block (4 wave32)

typedef __attribute__((ext_vector_type(16))) __bf16 v16bf;
typedef __attribute__((ext_vector_type(8)))  float  v8f;

struct BFrag { uint4 lo, hi; };

__device__ __forceinline__ v16bf pack_frag(uint4 lo, uint4 hi) {
  BFrag f{lo, hi};
  return __builtin_bit_cast(v16bf, f);
}

// A-matrix 16x32 bf16 fragment:
// lanes 0-15:  M=lane,    K = 0..15 of the 32-wide K block
// lanes 16-31: M=lane-16, K = 16..31
// 16 contiguous bf16 = two 16B loads.
__device__ __forceinline__ v16bf load_a_frag(const __bf16* base, int row0, int k0, int lane) {
  const int row = row0 + (lane & 15);
  const int kb  = k0 + ((lane >> 4) & 1) * 16;
  const __bf16* p = base + row * DMDIM + kb;
  return pack_frag(*(const uint4*)p, *(const uint4*)(p + 8));
}

// B-matrix 32x16 bf16 fragment: lane holds column n = n0 + lane%16,
// K = k0 + (lane/16)*16 .. +15, packed 2-per-VGPR in ascending K.
// With weights stored row-major W[n][k] this is 16 contiguous bf16 = two 16B loads.
__device__ __forceinline__ v16bf load_b_frag(const __bf16* W, int n0, int k0, int lane) {
  const int n = n0 + (lane & 15);
  const int k = k0 + ((lane >> 4) & 1) * 16;
  const __bf16* p = W + n * DMDIM + k;
  return pack_frag(*(const uint4*)p, *(const uint4*)(p + 8));
}

// [NNGH x DMDIM] = [32 x 384] @ W^T (W is [384 x 384] row-major) -> out [32 x 384] bf16.
// Each wave: 6 N-tiles x 2 M-tiles, K loop of 12 x 32.
__device__ __forceinline__ void proj_gemm_store(const __bf16* __restrict__ W,
                                                const __bf16* __restrict__ in,
                                                __bf16* __restrict__ out,
                                                int waveid, int lane) {
  const v8f vzero = {0.f, 0.f, 0.f, 0.f, 0.f, 0.f, 0.f, 0.f};
  v8f acc[2][6];
#pragma unroll
  for (int mt = 0; mt < 2; ++mt)
#pragma unroll
    for (int j = 0; j < 6; ++j) acc[mt][j] = vzero;

#pragma unroll 2
  for (int kb = 0; kb < DMDIM / 32; ++kb) {
    v16bf a0 = load_a_frag(in, 0,  kb * 32, lane);
    v16bf a1 = load_a_frag(in, 16, kb * 32, lane);
#pragma unroll
    for (int j = 0; j < 6; ++j) {
      const int n0 = (waveid * 6 + j) * 16;
      __builtin_prefetch(W + n0 * DMDIM + ((kb + 1) % (DMDIM / 32)) * 32, 0, 1);
      v16bf bm = load_b_frag(W, n0, kb * 32, lane);
      acc[0][j] = __builtin_amdgcn_wmma_f32_16x16x32_bf16(
          false, a0, false, bm, (short)0, acc[0][j], false, false);
      acc[1][j] = __builtin_amdgcn_wmma_f32_16x16x32_bf16(
          false, a1, false, bm, (short)0, acc[1][j], false, false);
    }
  }

  // C fragment: lane holds col n0 + lane%16; VGPR r holds row r + 8*(lane/16).
#pragma unroll
  for (int mt = 0; mt < 2; ++mt)
#pragma unroll
    for (int j = 0; j < 6; ++j) {
      const int n0 = (waveid * 6 + j) * 16 + (lane & 15);
      const int mb = mt * 16 + ((lane >> 4) & 1) * 8;
#pragma unroll
      for (int r = 0; r < 8; ++r)
        out[(mb + r) * DMDIM + n0] = (__bf16)acc[mt][j][r];
    }
}

// ---------------- weight bf16 conversion ----------------
__global__ void cvt_weights(const float* __restrict__ wk, const float* __restrict__ wv,
                            __bf16* __restrict__ wk_b, __bf16* __restrict__ wv_b, int n) {
  int i = blockIdx.x * blockDim.x + threadIdx.x;
  if (i < n) {
    wk_b[i] = (__bf16)wk[i];
    wv_b[i] = (__bf16)wv[i];
  }
}

// ---------------- fused attention + LN + merge ----------------
__launch_bounds__(NBLK, 1)
__global__ void tgat_attn_fused(
    const float* __restrict__ src,   const float* __restrict__ src_t,
    const float* __restrict__ seq,   const float* __restrict__ seq_t,
    const float* __restrict__ seq_e, const unsigned char* __restrict__ mask,
    const float* __restrict__ wq,
    const __bf16* __restrict__ wk_b, const __bf16* __restrict__ wv_b,
    const float* __restrict__ fc_w,   const float* __restrict__ fc_b,
    const float* __restrict__ ln_g,   const float* __restrict__ ln_b,
    const float* __restrict__ fc11_w, const float* __restrict__ fc11_b,
    const float* __restrict__ fc12_w, const float* __restrict__ fc12_b,
    const float* __restrict__ fc21_w, const float* __restrict__ fc21_b,
    const float* __restrict__ fc22_w, const float* __restrict__ fc22_b,
    float* __restrict__ out_merge, float* __restrict__ out_attn)
{
  __shared__ __align__(16) __bf16 inK[NNGH][DMDIM];   // staged concat rows (24 KB)
  __shared__ __align__(16) __bf16 khv[NNGH][DMDIM];   // kh then vh          (24 KB)
  __shared__ float qv[DMDIM], qh[DMDIM], outv[DMDIM], res[DMDIM];
  __shared__ float sc[NHEAD][NNGH], attnS[NHEAD][NNGH];
  __shared__ float red1[NBLK], red2[NBLK];
  __shared__ float h1[FEAT], h2[FEAT];
  __shared__ float s_mu, s_rstd;

  const int tid    = threadIdx.x;
  const int lane   = tid & 31;
  const int waveid = tid >> 5;
  const int bs     = blockIdx.x;       // b*NSRC + s
  const int b      = bs >> 4;
  const int s      = bs & 15;

  // ---- stage neighbor rows (feat|edge|time) as bf16; build q vector ----
  for (int idx = tid; idx < NNGH * DMDIM; idx += NBLK) {
    const int n = idx / DMDIM, c = idx % DMDIM;
    const int gj = b * NNTOT + s * NNGH + n;
    float v;
    if (c < FEAT)              v = seq  [gj * FEAT  + c];
    else if (c < FEAT + EDGEF) v = seq_e[gj * EDGEF + (c - FEAT)];
    else                       v = seq_t[gj * TIMEF + (c - FEAT - EDGEF)];
    inK[n][c] = (__bf16)v;
  }
  for (int c = tid; c < DMDIM; c += NBLK) {
    float v = 0.f;                      // edge slice of q is zero
    if (c < FEAT)               v = src  [bs * FEAT  + c];
    else if (c >= FEAT + EDGEF) v = src_t[bs * TIMEF + (c - FEAT - EDGEF)];
    qv[c] = v;
  }
  __syncthreads();

  // ---- q projection (f32 GEMV; wq L2-resident) ----
  for (int j = tid; j < DMDIM; j += NBLK) {
    const float* wr = wq + j * DMDIM;
    float acc = 0.f;
    for (int k = 0; k < FEAT; ++k)             acc += qv[k] * wr[k];
    for (int k = FEAT + EDGEF; k < DMDIM; ++k) acc += qv[k] * wr[k];
    qh[j] = acc;
  }

  // ---- K projection: WMMA bf16 ----
  proj_gemm_store(wk_b, &inK[0][0], &khv[0][0], waveid, lane);
  __syncthreads();

  // ---- scores + faithful mask replication ----
  if (tid < NHEAD * NNGH) {
    const int h = tid >> 5, n = tid & 31;
    const float*  q0 = qh + h * DKDIM;
    const __bf16* kr = &khv[n][h * DKDIM];
    float acc = 0.f;
    for (int d = 0; d < DKDIM; ++d) acc += q0[d] * (float)kr[d];
    acc *= 0.07216878364870323f;     // 1/sqrt(192)
    const int mrow = (bs * NHEAD + h) & (BATCH * NSRC - 1);  // % 8192
    if (mask[mrow * NNGH + n]) acc = -1e10f;
    sc[h][n] = acc;
  }
  __syncthreads();

  // ---- softmax per head over 32 neighbors ----
  if (tid < NHEAD) {
    const int h = tid;
    float mx = -3.4e38f;
    for (int n = 0; n < NNGH; ++n) mx = fmaxf(mx, sc[h][n]);
    float e[NNGH], sum = 0.f;
    for (int n = 0; n < NNGH; ++n) { e[n] = __expf(sc[h][n] - mx); sum += e[n]; }
    const float inv = 1.f / sum;
    for (int n = 0; n < NNGH; ++n) {
      const float a = e[n] * inv;
      attnS[h][n] = a;
      out_attn[(size_t)bs * (NHEAD * NNGH) + h * NNGH + n] = a;
    }
  }
  __syncthreads();

  // ---- V projection: WMMA bf16 (kh buffer now dead -> reuse) ----
  proj_gemm_store(wv_b, &inK[0][0], &khv[0][0], waveid, lane);
  __syncthreads();

  // ---- out = attn @ V ----
  for (int c = tid; c < DMDIM; c += NBLK) {
    const int h = c / DKDIM;
    float acc = 0.f;
    for (int n = 0; n < NNGH; ++n) acc += attnS[h][n] * (float)khv[n][c];
    outv[c] = acc;
  }
  __syncthreads();

  // ---- fc + bias + residual ----
  for (int c = tid; c < DMDIM; c += NBLK) {
    const float* wr = fc_w + c * DMDIM;
    float acc = fc_b[c] + qv[c];
    for (int k = 0; k < DMDIM; ++k) acc += outv[k] * wr[k];
    res[c] = acc;
  }
  __syncthreads();

  // ---- LayerNorm over 384 ----
  {
    float p1 = 0.f, p2 = 0.f;
    for (int c = tid; c < DMDIM; c += NBLK) { float x = res[c]; p1 += x; p2 += x * x; }
    red1[tid] = p1; red2[tid] = p2;
    __syncthreads();
    if (tid == 0) {
      float s1 = 0.f, s2 = 0.f;
      for (int i = 0; i < NBLK; ++i) { s1 += red1[i]; s2 += red2[i]; }
      const float mu = s1 / DMDIM;
      s_mu = mu;
      s_rstd = rsqrtf(s2 / DMDIM - mu * mu + 1e-5f);
    }
    __syncthreads();
    for (int c = tid; c < DMDIM; c += NBLK)
      outv[c] = (res[c] - s_mu) * s_rstd * ln_g[c] + ln_b[c];
  }
  __syncthreads();

  // ---- MergeLayer: relu branches ----
  {
    const int j = tid;             // NBLK == FEAT == 128
    float a1 = fc11_b[j];
    const float* w1 = fc11_w + j * DMDIM;
    for (int k = 0; k < DMDIM; ++k) a1 += outv[k] * w1[k];
    h1[j] = fmaxf(a1, 0.f);
    float a2 = fc12_b[j];
    const float* w2 = fc12_w + j * FEAT;
    for (int k = 0; k < FEAT; ++k) a2 += qv[k] * w2[k];
    h2[j] = fmaxf(a2, 0.f);
  }
  __syncthreads();
  {
    const int j = tid;
    float acc = fc21_b[j] + fc22_b[j];
    const float* w1 = fc21_w + j * FEAT;
    const float* w2 = fc22_w + j * FEAT;
    for (int k = 0; k < FEAT; ++k) acc += h1[k] * w1[k] + h2[k] * w2[k];
    out_merge[(size_t)bs * FEAT + j] = acc;
  }
}

// ---------------- host-side launcher ----------------
extern "C" void kernel_launch(void* const* d_in, const int* in_sizes, int n_in,
                              void* d_out, int out_size, void* d_ws, size_t ws_size,
                              hipStream_t stream) {
  (void)in_sizes; (void)n_in; (void)out_size; (void)ws_size;

  const float* src    = (const float*)d_in[0];
  const float* src_t  = (const float*)d_in[1];
  const float* seq    = (const float*)d_in[2];
  const float* seq_t  = (const float*)d_in[3];
  const float* seq_e  = (const float*)d_in[4];
  const unsigned char* mask = (const unsigned char*)d_in[5];
  const float* wq     = (const float*)d_in[6];
  const float* wk     = (const float*)d_in[7];
  const float* wv     = (const float*)d_in[8];
  const float* fc_w   = (const float*)d_in[9];
  const float* fc_b   = (const float*)d_in[10];
  const float* ln_g   = (const float*)d_in[11];
  const float* ln_b   = (const float*)d_in[12];
  const float* fc11_w = (const float*)d_in[13];
  const float* fc11_b = (const float*)d_in[14];
  const float* fc12_w = (const float*)d_in[15];
  const float* fc12_b = (const float*)d_in[16];
  const float* fc21_w = (const float*)d_in[17];
  const float* fc21_b = (const float*)d_in[18];
  const float* fc22_w = (const float*)d_in[19];
  const float* fc22_b = (const float*)d_in[20];

  __bf16* wk_b = (__bf16*)d_ws;
  __bf16* wv_b = wk_b + DMDIM * DMDIM;

  const int nw = DMDIM * DMDIM;
  cvt_weights<<<(nw + 255) / 256, 256, 0, stream>>>(wk, wv, wk_b, wv_b, nw);

  float* out_merge = (float*)d_out;                                   // [8192,128]
  float* out_attn  = out_merge + (size_t)BATCH * NSRC * FEAT;         // [8192,2,32]

  tgat_attn_fused<<<BATCH * NSRC, NBLK, 0, stream>>>(
      src, src_t, seq, seq_t, seq_e, mask, wq, wk_b, wv_b,
      fc_w, fc_b, ln_g, ln_b, fc11_w, fc11_b, fc12_w, fc12_b,
      fc21_w, fc21_b, fc22_w, fc22_b, out_merge, out_attn);
}
